// CapsuleLayer_58523224375783
// MI455X (gfx1250) — matrix operand order repaired and checked by reading, not compiled, not run on hardware
//
#include <hip/hip_runtime.h>

// ---------------------------------------------------------------------------
// Types
// ---------------------------------------------------------------------------
typedef __attribute__((ext_vector_type(16))) __bf16   v16bf;
typedef __attribute__((ext_vector_type(8)))  float    v8f;
typedef __attribute__((ext_vector_type(4)))  float    float4v;
typedef __attribute__((ext_vector_type(2)))  unsigned uint2v;
typedef __attribute__((ext_vector_type(4)))  unsigned uint4v;
typedef __attribute__((ext_vector_type(4)))  int      int4v;

// address-space qualified int4 for the async-to-LDS builtin
typedef __attribute__((address_space(1))) int4v as1_int4;
typedef __attribute__((address_space(3))) int4v as3_int4;

struct F8 { uint2v a, b, c, d; };  // 32 bytes == v16bf

static __device__ inline unsigned short f2bf(float f) {
    unsigned u = __builtin_bit_cast(unsigned, f);
    unsigned r = 0x7FFFu + ((u >> 16) & 1u);            // round-to-nearest-even
    return (unsigned short)((u + r) >> 16);
}
static __device__ inline float bf2f(unsigned short h) {
    return __builtin_bit_cast(float, ((unsigned)h) << 16);
}

#define LDS_S 40   // bf16 row stride: 80B rows -> 16B-aligned staging, conflict-free frags

// ---------------------------------------------------------------------------
// Async global->LDS staging (CDNA5 ASYNCcnt path) with compile-safe fallback
// ---------------------------------------------------------------------------
#if defined(__has_builtin)
#  if __has_builtin(__builtin_amdgcn_global_load_async_to_lds_b128)
#    define USE_ASYNC_LDS 1
#  endif
#endif
#ifndef USE_ASYNC_LDS
#  define USE_ASYNC_LDS 0
#endif

static __device__ inline void copy16(const unsigned short* g, unsigned short* l) {
#if USE_ASYNC_LDS
    __builtin_amdgcn_global_load_async_to_lds_b128(
        (as1_int4*)g, (as3_int4*)l, 0, 0);
#else
    *(uint4v*)l = *(const uint4v*)g;
#endif
}
static __device__ inline void zero16(unsigned short* l) {
    uint4v z = {0u, 0u, 0u, 0u};
    *(uint4v*)l = z;
}
static __device__ inline void wait_stage() {
#if USE_ASYNC_LDS
#  if __has_builtin(__builtin_amdgcn_s_wait_asynccnt)
    __builtin_amdgcn_s_wait_asynccnt(0);
#  else
    asm volatile("s_wait_asynccnt 0" ::: "memory");
#  endif
#endif
}

// ---------------------------------------------------------------------------
// WMMA fragment loads from LDS (bf16 planes)
// ---------------------------------------------------------------------------
// A fragment (16x32, ISA K-interleave): lanes 0-15 kgrp=0: V0..3=K0..7, V4..7=K16..23
static __device__ inline v16bf load_fragA(const unsigned short* s, int row, int kgrp) {
    const int base = row * LDS_S + kgrp * 8;
    F8 f;
    f.a = *(const uint2v*)(s + base);
    f.b = *(const uint2v*)(s + base + 4);
    f.c = *(const uint2v*)(s + base + 16);
    f.d = *(const uint2v*)(s + base + 20);
    return __builtin_bit_cast(v16bf, f);
}
// B fragment (32x16, ISA K-interleave): lanes 0-15 kset=0: V0..3=K0..7, V4..7=K16..23
static __device__ inline v16bf load_fragB(const unsigned short* s, int row, int kset) {
    const int base = row * LDS_S + kset * 8;
    F8 f;
    f.a = *(const uint2v*)(s + base);
    f.b = *(const uint2v*)(s + base + 4);
    f.c = *(const uint2v*)(s + base + 16);
    f.d = *(const uint2v*)(s + base + 20);
    return __builtin_bit_cast(v16bf, f);
}
static __device__ inline v8f wmma_bf16(v16bf a, v16bf b, v8f c) {
    return __builtin_amdgcn_wmma_f32_16x16x32_bf16(
        false, a, false, b, (short)0, c, false, false);
}

// ---------------------------------------------------------------------------
// Split fp32 -> bf16 hi/lo planes (same layout).  n multiple of 4.
// ---------------------------------------------------------------------------
__global__ __launch_bounds__(256) void split_plane(
    const float* __restrict__ src, unsigned short* __restrict__ hi,
    unsigned short* __restrict__ lo, size_t n)
{
    size_t stride = (size_t)gridDim.x * 256 * 4;
    for (size_t i = ((size_t)blockIdx.x * 256 + threadIdx.x) * 4; i < n; i += stride) {
        float4v v = *(const float4v*)(src + i);
        unsigned short h0 = f2bf(v.x), h1 = f2bf(v.y), h2 = f2bf(v.z), h3 = f2bf(v.w);
        unsigned short l0 = f2bf(v.x - bf2f(h0)), l1 = f2bf(v.y - bf2f(h1));
        unsigned short l2 = f2bf(v.z - bf2f(h2)), l3 = f2bf(v.w - bf2f(h3));
        uint2v hv = { (unsigned)h0 | ((unsigned)h1 << 16), (unsigned)h2 | ((unsigned)h3 << 16) };
        uint2v lv = { (unsigned)l0 | ((unsigned)l1 << 16), (unsigned)l2 | ((unsigned)l3 << 16) };
        *(uint2v*)(hi + i) = hv;
        *(uint2v*)(lo + i) = lv;
    }
}

// ---------------------------------------------------------------------------
// Split + transpose weights: src[n][k][j] fp32 -> dst[n][j][k] bf16 hi/lo
// block (32,8), 32x32 tiles. grid = (ceil(J/32), ceil(K/32), N)
// ---------------------------------------------------------------------------
__global__ __launch_bounds__(256) void split_transpose(
    const float* __restrict__ src, size_t strideSrcN, size_t ldSrc,
    int K, int J,
    unsigned short* __restrict__ hi, unsigned short* __restrict__ lo)
{
    __shared__ float t[32][33];
    const int n  = blockIdx.z;
    const int j0 = blockIdx.x * 32;
    const int k0 = blockIdx.y * 32;
    const int tx = threadIdx.x, ty = threadIdx.y;
    #pragma unroll
    for (int r = 0; r < 4; ++r) {
        int k = k0 + ty + r * 8, j = j0 + tx;
        t[ty + r * 8][tx] = (k < K && j < J)
            ? src[(size_t)n * strideSrcN + (size_t)k * ldSrc + (size_t)j] : 0.0f;
    }
    __syncthreads();
    #pragma unroll
    for (int r = 0; r < 4; ++r) {
        int j = j0 + ty + r * 8, k = k0 + tx;
        if (j < J && k < K) {
            float v = t[tx][ty + r * 8];
            unsigned short h = f2bf(v);
            size_t o = ((size_t)n * (size_t)J + (size_t)j) * (size_t)K + (size_t)k;
            hi[o] = h;
            lo[o] = f2bf(v - bf2f(h));
        }
    }
}

// ---------------------------------------------------------------------------
// Per-capsule batched GEMM, bf16x3 split precision, pre-split operand planes.
//   C[n][m][j] = act( sum_k A[n][m][k]*B[n][k][j] + bias[n][j] )
// A planes: [.. m*ldA + k] (k contiguous). B planes: Bt [n][J][K] (k contiguous).
// 256 threads per 64x64 C tile; double-buffered LDS; grid (M/64, ceil(J/64), N)
// ---------------------------------------------------------------------------
__global__ __launch_bounds__(256) void gemm_caps_planes(
    const unsigned short* __restrict__ Ahi, const unsigned short* __restrict__ Alo,
    size_t ldA, size_t strideAn,
    const unsigned short* __restrict__ Bhi, const unsigned short* __restrict__ Blo,
    size_t strideBn,
    const float* __restrict__ bias, size_t strideBias,
    float* __restrict__ Cf,                               // optional fp32 output
    unsigned short* __restrict__ Chi, unsigned short* __restrict__ Clo, // optional planes
    size_t ldC, size_t strideCn,
    int K, int Jtot, int doRelu)
{
    __shared__ unsigned short sAh[2][64 * LDS_S];
    __shared__ unsigned short sAl[2][64 * LDS_S];
    __shared__ unsigned short sBh[2][64 * LDS_S];
    __shared__ unsigned short sBl[2][64 * LDS_S];

    const int tid  = threadIdx.x;
    const int lane = tid & 31;
    const int wave = tid >> 5;
    const int n    = blockIdx.z;
    const int m0   = blockIdx.x * 64;
    const int j0   = blockIdx.y * 64;

    // staging: each thread moves 16B (8 bf16) per plane per K-step
    const int rowT = tid >> 2;          // 0..63
    const int segT = (tid & 3) * 8;     // 0,8,16,24
    const unsigned short* gAh = Ahi + (size_t)n * strideAn + (size_t)(m0 + rowT) * ldA + segT;
    const unsigned short* gAl = Alo + (size_t)n * strideAn + (size_t)(m0 + rowT) * ldA + segT;
    const int  jg  = j0 + rowT;
    const bool bok = (jg < Jtot);
    const size_t bOff = (size_t)n * strideBn + (size_t)jg * (size_t)K + segT;
    const unsigned short* gBh = Bhi + bOff;
    const unsigned short* gBl = Blo + bOff;
    const int ldsOff = rowT * LDS_S + segT;

    auto stage = [&](int buf, int k0) {
        copy16(gAh + k0, &sAh[buf][ldsOff]);
        copy16(gAl + k0, &sAl[buf][ldsOff]);
        if (bok) {
            copy16(gBh + k0, &sBh[buf][ldsOff]);
            copy16(gBl + k0, &sBl[buf][ldsOff]);
        } else {
            zero16(&sBh[buf][ldsOff]);
            zero16(&sBl[buf][ldsOff]);
        }
    };

    // wave tiling: 4 row tiles x 2 col groups (each 2 x 16-wide tiles)
    const int rtile = wave & 3;
    const int cgrp  = wave >> 2;
    const int mfrag = rtile * 16 + (lane & 15);
    const int kgrp  = lane >> 4;
    const int jrow0 = cgrp * 32 + (lane & 15);
    const int jrow1 = jrow0 + 16;

    v8f acc0 = {};
    v8f acc1 = {};

    int cur = 0;
    stage(0, 0);
    for (int k0 = 0; k0 < K; k0 += 32) {
        wait_stage();
        __syncthreads();
        if (k0 + 32 < K) stage(cur ^ 1, k0 + 32);

        v16bf ah  = load_fragA(sAh[cur], mfrag, kgrp);
        v16bf al  = load_fragA(sAl[cur], mfrag, kgrp);
        v16bf bh0 = load_fragB(sBh[cur], jrow0, kgrp);
        v16bf bl0 = load_fragB(sBl[cur], jrow0, kgrp);
        v16bf bh1 = load_fragB(sBh[cur], jrow1, kgrp);
        v16bf bl1 = load_fragB(sBl[cur], jrow1, kgrp);

        acc0 = wmma_bf16(ah, bh0, acc0);
        acc0 = wmma_bf16(ah, bl0, acc0);
        acc0 = wmma_bf16(al, bh0, acc0);
        acc1 = wmma_bf16(ah, bh1, acc1);
        acc1 = wmma_bf16(ah, bl1, acc1);
        acc1 = wmma_bf16(al, bh1, acc1);
        cur ^= 1;
    }

    // epilogue: bias + relu; write fp32 C and/or bf16 hi/lo planes
    const int jc0 = j0 + cgrp * 32 + (lane & 15);
    const int jc1 = jc0 + 16;
    float b0 = 0.0f, b1 = 0.0f;
    if (bias) {
        if (jc0 < Jtot) b0 = bias[(size_t)n * strideBias + (size_t)jc0];
        if (jc1 < Jtot) b1 = bias[(size_t)n * strideBias + (size_t)jc1];
    }
    const size_t cBase = (size_t)n * strideCn;
    #pragma unroll
    for (int vv = 0; vv < 8; ++vv) {
        int m = m0 + rtile * 16 + vv + (lane >> 4) * 8;
        size_t o0 = cBase + (size_t)m * ldC + (size_t)jc0;
        size_t o1 = cBase + (size_t)m * ldC + (size_t)jc1;
        float x0 = acc0[vv] + b0, x1 = acc1[vv] + b1;
        if (doRelu) { x0 = fmaxf(x0, 0.0f); x1 = fmaxf(x1, 0.0f); }
        if (jc0 < Jtot) {
            if (Cf)  Cf[o0] = x0;
            if (Chi) { unsigned short h = f2bf(x0); Chi[o0] = h; Clo[o0] = f2bf(x0 - bf2f(h)); }
        }
        if (jc1 < Jtot) {
            if (Cf)  Cf[o1] = x1;
            if (Chi) { unsigned short h = f2bf(x1); Chi[o1] = h; Clo[o1] = f2bf(x1 - bf2f(h)); }
        }
    }
}

// ---------------------------------------------------------------------------
// Elementwise tail: transforms, vote, presence, scale, reg loss
// ---------------------------------------------------------------------------
static __device__ inline float sigm(float x)     { return 1.0f / (1.0f + expf(-x)); }
static __device__ inline float softplusf(float x){ return fmaxf(x, 0.0f) + log1pf(expf(-fabsf(x))); }

static __device__ inline void make_t(const float p[6], float M[9]) {
    float scale = sigm(p[0]) + 1e-2f;
    float tx = tanhf(p[4] * 5.0f);
    float ty = tanhf(p[5] * 5.0f);
    float th = p[2] * 6.283185307179586f;
    float s, c;
    sincosf(th, &s, &c);
    M[0] = scale * c;  M[1] = -scale * s; M[2] = tx;
    M[3] = scale * s;  M[4] =  scale * c; M[5] = ty;
    M[6] = 0.0f;       M[7] = 0.0f;       M[8] = 1.0f;
}

__global__ __launch_bounds__(256) void caps_post(
    const float* __restrict__ ap,
    const float* __restrict__ bias_cvr,
    const float* __restrict__ bias_pl_caps,
    const float* __restrict__ bias_pl_vote,
    const float* __restrict__ bias_scale,
    const float* __restrict__ cpr_static,
    float* __restrict__ out_vote, float* __restrict__ out_scale,
    float* __restrict__ out_pres, float* __restrict__ out_plc,
    float* __restrict__ out_plv,  float* __restrict__ out_loss)
{
    const int idx = blockIdx.x * 256 + threadIdx.x;   // b*N*16 + n*16 + v
    const int v  = idx & 15;
    const int bn = idx >> 4;
    const int n  = bn & 63;
    const float* p = ap + (size_t)bn * 135;

    float reg = 0.0f;
    float c6[6], q6[6];
    #pragma unroll
    for (int t = 0; t < 6; ++t) {
        float d = p[v * 6 + t];
        reg += d * d;
        c6[t] = d + cpr_static[(size_t)(n * 16 + v) * 6 + t];
        q6[t] = p[96 + t] + bias_cvr[n * 6 + t];
    }

    float P[9], Q[9], Vt[9];
    make_t(c6, P);
    make_t(q6, Q);
    #pragma unroll
    for (int r = 0; r < 3; ++r)
        #pragma unroll
        for (int cc = 0; cc < 3; ++cc) {
            float s = 0.0f;
            #pragma unroll
            for (int k2 = 0; k2 < 3; ++k2) s += Q[r * 3 + k2] * P[k2 * 3 + cc];
            Vt[r * 3 + cc] = s;
        }
    float* vo = out_vote + (size_t)idx * 9;
    #pragma unroll
    for (int e = 0; e < 9; ++e) vo[e] = Vt[e];

    float plc = p[102]     + bias_pl_caps[n];
    float plv = p[103 + v] + bias_pl_vote[n * 16 + v];
    out_plv[idx]  = plv;
    if (v == 0) out_plc[bn] = plc;
    out_pres[idx]  = sigm(plc) * sigm(plv);
    out_scale[idx] = softplusf(p[119 + v] + bias_scale[n * 16 + v] + 0.5f) + 0.01f;

    reg *= (0.5f / 1024.0f);
    #pragma unroll
    for (int off = 16; off > 0; off >>= 1) reg += __shfl_down(reg, off, 32);
    if ((threadIdx.x & 31) == 0) atomicAdd(out_loss, reg);
}

__global__ void zero1(float* p) { *p = 0.0f; }

// ---------------------------------------------------------------------------
// Launch: B=1024, N=64, F=512, H=512, D=64, V=16, NOUT=135
// ---------------------------------------------------------------------------
extern "C" void kernel_launch(void* const* d_in, const int* in_sizes, int n_in,
                              void* d_out, int out_size, void* d_ws, size_t ws_size,
                              hipStream_t stream) {
    const float* feature      = (const float*)d_in[0];
    const float* mlp_w1       = (const float*)d_in[1];
    const float* mlp_b1       = (const float*)d_in[2];
    const float* mlp_w2       = (const float*)d_in[3];
    const float* mlp_b2       = (const float*)d_in[4];
    const float* caps_w1      = (const float*)d_in[5];
    const float* caps_w2      = (const float*)d_in[6];
    const float* bias_cvr     = (const float*)d_in[7];
    const float* bias_pl_caps = (const float*)d_in[8];
    const float* bias_pl_vote = (const float*)d_in[9];
    const float* bias_scale   = (const float*)d_in[10];
    const float* cpr_static   = (const float*)d_in[11];
    float* out = (float*)d_out;

    // output layout (concatenated, return order)
    const size_t VOTE_OFF  = 0;
    const size_t SCALE_OFF = 9437184;
    const size_t PRES_OFF  = SCALE_OFF + 1048576;
    const size_t PLC_OFF   = PRES_OFF  + 1048576;
    const size_t PLV_OFF   = PLC_OFF   + 65536;
    const size_t LOSS_OFF  = PLV_OFF   + 1048576;
    const size_t RAW_OFF   = LOSS_OFF  + 1;

    // workspace carve (bf16 planes + fp32 all_param)
    unsigned short* w = (unsigned short*)d_ws;
    const size_t FEAT_N = (size_t)1024 * 64 * 512;   // 33.55M
    const size_t H_N    = (size_t)1024 * 64 * 512;
    const size_t RAW_N  = (size_t)1024 * 64 * 64;
    const size_t W1T_N  = (size_t)64 * 512 * 512;
    const size_t W2T_N  = (size_t)64 * 64 * 512;
    const size_t CW1T_N = (size_t)64 * 512 * 64;
    const size_t CW2T_N = (size_t)64 * 135 * 512;

    unsigned short* featH = w;            w += FEAT_N;
    unsigned short* featL = w;            w += FEAT_N;
    unsigned short* hH    = w;            w += H_N;    // reused for h2 planes
    unsigned short* hL    = w;            w += H_N;
    unsigned short* rawH  = w;            w += RAW_N;
    unsigned short* rawL  = w;            w += RAW_N;
    unsigned short* w1tH  = w;            w += W1T_N;
    unsigned short* w1tL  = w;            w += W1T_N;
    unsigned short* w2tH  = w;            w += W2T_N;
    unsigned short* w2tL  = w;            w += W2T_N;
    unsigned short* cw1tH = w;            w += CW1T_N;
    unsigned short* cw1tL = w;            w += CW1T_N;
    unsigned short* cw2tH = w;            w += CW2T_N;
    unsigned short* cw2tL = w;            w += CW2T_N;
    float* apw = (float*)w;
    float* raw = out + RAW_OFF;

    dim3 blk(256);
    dim3 tblk(32, 8);

    zero1<<<1, 1, 0, stream>>>(out + LOSS_OFF);

    // preconversion passes
    split_plane<<<8192, blk, 0, stream>>>(feature, featH, featL, FEAT_N);
    split_transpose<<<dim3(16, 16, 64), tblk, 0, stream>>>(
        mlp_w1, (size_t)512 * 512, 512, 512, 512, w1tH, w1tL);
    split_transpose<<<dim3(2, 16, 64), tblk, 0, stream>>>(
        mlp_w2, (size_t)512 * 64, 64, 512, 64, w2tH, w2tL);
    split_transpose<<<dim3(16, 2, 64), tblk, 0, stream>>>(
        caps_w1, (size_t)65 * 512, 512, 64, 512, cw1tH, cw1tL);   // first 64 K rows
    split_transpose<<<dim3(5, 16, 64), tblk, 0, stream>>>(
        caps_w2, (size_t)512 * 135, 135, 512, 135, cw2tH, cw2tL);

    // G1: h = relu(feature @ mlp_w1 + b1) -> planes only
    gemm_caps_planes<<<dim3(16, 8, 64), blk, 0, stream>>>(
        featH, featL, (size_t)64 * 512, (size_t)512,
        w1tH, w1tL, (size_t)512 * 512,
        mlp_b1, (size_t)512,
        nullptr, hH, hL, (size_t)64 * 512, (size_t)512,
        512, 512, 1);

    // G2: raw = h @ mlp_w2 + b2 -> fp32 (output) + planes
    gemm_caps_planes<<<dim3(16, 1, 64), blk, 0, stream>>>(
        hH, hL, (size_t)64 * 512, (size_t)512,
        w2tH, w2tL, (size_t)64 * 512,
        mlp_b2, (size_t)64,
        raw, rawH, rawL, (size_t)64 * 64, (size_t)64,
        512, 64, 0);

    // G3: h2 = relu(raw @ caps_w1[:, :64, :] + caps_w1[:, 64, :]) -> planes (reuse h)
    gemm_caps_planes<<<dim3(16, 8, 64), blk, 0, stream>>>(
        rawH, rawL, (size_t)64 * 64, (size_t)64,
        cw1tH, cw1tL, (size_t)512 * 64,
        caps_w1 + 64 * 512, (size_t)65 * 512,
        nullptr, hH, hL, (size_t)64 * 512, (size_t)512,
        64, 512, 1);

    // G4: all_param = h2 @ caps_w2 -> fp32, J=135 padded to 192
    gemm_caps_planes<<<dim3(16, 3, 64), blk, 0, stream>>>(
        hH, hL, (size_t)64 * 512, (size_t)512,
        cw2tH, cw2tL, (size_t)135 * 512,
        nullptr, (size_t)0,
        apw, nullptr, nullptr, (size_t)64 * 135, (size_t)135,
        512, 135, 0);

    // tail
    caps_post<<<dim3((1024 * 64 * 16) / 256), blk, 0, stream>>>(
        apw, bias_cvr, bias_pl_caps, bias_pl_vote, bias_scale, cpr_static,
        out + VOTE_OFF, out + SCALE_OFF, out + PRES_OFF,
        out + PLC_OFF, out + PLV_OFF, out + LOSS_OFF);
}
